// GCS_cls_25486335935326
// MI455X (gfx1250) — matrix operand order repaired and checked by reading, not compiled, not run on hardware
//
#include <hip/hip_runtime.h>

typedef __attribute__((ext_vector_type(16))) _Float16 v16h;
typedef __attribute__((ext_vector_type(8)))  float    v8f;

#define BN_EPS 1e-5f
#define BATCH 128

// ---------------------------------------------------------------------------
// WMMA GEMM: C[M,N] = epi( A[M,K] @ W[K,N] ),  M % 64 == 0 (3072 / 128 here).
// One wave computes a 64x16 tile: 4 A-fragments share one B-fragment ->
// 4 v_wmma_f32_16x16x32_f16 per K-step. No predicated loads: fast path is
// unconditional float4/b32 clauses; tails use clamped addresses + cndmask.
// epi: (+bias[n]) -> BN(n) -> relu -> (+resid[m,n])
// ---------------------------------------------------------------------------
__global__ __launch_bounds__(32)
void wmma_gemm_kernel(const float* __restrict__ A, const float* __restrict__ W,
                      float* __restrict__ C, int M, int N, int K,
                      const float* __restrict__ bias,
                      const float* __restrict__ bng, const float* __restrict__ bnb,
                      const float* __restrict__ bnm, const float* __restrict__ bnv,
                      const float* __restrict__ resid, int relu)
{
    const int lane = threadIdx.x;      // 0..31
    const int half = lane >> 4;        // 0|1
    const int lr   = lane & 15;
    const int tm   = blockIdx.y * 64;
    const int tn   = blockIdx.x * 16;
    const int n    = tn + lr;          // B/C column owned by this lane
    const bool n_ok = (n < N);
    const int  n_cl = n_ok ? n : (N - 1);

    v8f acc[4] = {{}, {}, {}, {}};

    for (int k0 = 0; k0 < K; k0 += 32) {
        v16h b;
        if (k0 + 32 <= K) {            // uniform scalar branch (EXEC untouched)
            // ---- B 32x16 fragment: lane = column, elem t -> K = 8*half + t (+8 for t>=8)
            const float* wp = W + (long)(k0 + half * 8) * N + n_cl;
            #pragma unroll
            for (int t = 0; t < 16; ++t) {
                float wv = wp[(long)(t + (t & 8)) * N];            // unconditional, coalesced
                b[t] = n_ok ? (_Float16)wv : (_Float16)0.f;
            }
            // ---- 4 A fragments (16x32 each), vectorized b128 loads
            #pragma unroll
            for (int i = 0; i < 4; ++i) {
                const float* ar = A + (long)(tm + i * 16 + lr) * K + k0 + half * 8;
                float4 p0 = *(const float4*)(ar);
                float4 p1 = *(const float4*)(ar + 4);
                float4 p2 = *(const float4*)(ar + 16);
                float4 p3 = *(const float4*)(ar + 20);
                v16h a;
                a[0]  = (_Float16)p0.x; a[1]  = (_Float16)p0.y;
                a[2]  = (_Float16)p0.z; a[3]  = (_Float16)p0.w;
                a[4]  = (_Float16)p1.x; a[5]  = (_Float16)p1.y;
                a[6]  = (_Float16)p1.z; a[7]  = (_Float16)p1.w;
                a[8]  = (_Float16)p2.x; a[9]  = (_Float16)p2.y;
                a[10] = (_Float16)p2.z; a[11] = (_Float16)p2.w;
                a[12] = (_Float16)p3.x; a[13] = (_Float16)p3.y;
                a[14] = (_Float16)p3.z; a[15] = (_Float16)p3.w;
                acc[i] = __builtin_amdgcn_wmma_f32_16x16x32_f16(false, a, false, b,
                                                               (short)0, acc[i], false, false);
            }
        } else {                        // K tail (K = 24 or 48 only): clamp + select
            #pragma unroll
            for (int t = 0; t < 16; ++t) {
                int kk  = k0 + half * 8 + t + (t & 8);
                int kcl = kk < K ? kk : K - 1;
                float wv = W[(long)kcl * N + n_cl];    // clamped, unconditional
                b[t] = (n_ok && kk < K) ? (_Float16)wv : (_Float16)0.f;
            }
            #pragma unroll
            for (int i = 0; i < 4; ++i) {
                const float* ar = A + (long)(tm + i * 16 + lr) * K;
                v16h a;
                #pragma unroll
                for (int t = 0; t < 8; ++t) {
                    int k1 = k0 + half * 8 + t;
                    int k2 = k1 + 16;
                    int c1 = k1 < K ? k1 : K - 1;
                    int c2 = k2 < K ? k2 : K - 1;
                    float av1 = ar[c1], av2 = ar[c2];  // clamped, unconditional
                    a[t]     = (k1 < K) ? (_Float16)av1 : (_Float16)0.f;
                    a[t + 8] = (k2 < K) ? (_Float16)av2 : (_Float16)0.f;
                }
                acc[i] = __builtin_amdgcn_wmma_f32_16x16x32_f16(false, a, false, b,
                                                               (short)0, acc[i], false, false);
            }
        }
    }

    // ---- epilogue: n is fixed per lane -> hoist per-column params
    float bias_v = bias ? bias[n_cl] : 0.f;
    float scale = 1.f, shift = 0.f;
    if (bng) {
        scale = bng[n_cl] * rsqrtf(bnv[n_cl] + BN_EPS);
        shift = bnb[n_cl] - bnm[n_cl] * scale;
    }
    #pragma unroll
    for (int i = 0; i < 4; ++i) {
        #pragma unroll
        for (int r = 0; r < 8; ++r) {
            int mo = tm + i * 16 + half * 8 + r;       // C layout: VGPR r -> row half*8+r
            float v = acc[i][r] + bias_v;
            if (bng)  v = v * scale + shift;
            if (relu) v = fmaxf(v, 0.f);
            if (n_ok) {
                if (resid) v += resid[(long)mo * N + n];
                C[(long)mo * N + n] = v;
            }
        }
    }
}

// ---------------------------------------------------------------------------
// Slice x[..., :216] -> [B,1,384,216]
// ---------------------------------------------------------------------------
__global__ void slice_kernel(const float* __restrict__ x, float* __restrict__ out)
{
    long idx = (long)blockIdx.x * blockDim.x + threadIdx.x;
    long total = (long)BATCH * 384 * 216;
    if (idx >= total) return;
    int w = idx % 216; long r = idx / 216;   // r = b*384 + h
    out[idx] = x[r * 400 + w];
}

// ---------------------------------------------------------------------------
// Direct conv3x3 (pad 1) + BN2d (+ ReLU). Small channel counts -> VALU f32.
// ---------------------------------------------------------------------------
__global__ void conv_bn_kernel(const float* __restrict__ in, float* __restrict__ out,
                               const float* __restrict__ w, const float* __restrict__ bias,
                               const float* __restrict__ g, const float* __restrict__ be,
                               const float* __restrict__ mu, const float* __restrict__ var,
                               int Ci, int Co, int Hi, int Wi, int Ho, int Wo,
                               int sh, int sw, int relu)
{
    long idx = (long)blockIdx.x * blockDim.x + threadIdx.x;
    long total = (long)BATCH * Co * Ho * Wo;
    if (idx >= total) return;
    int wo = idx % Wo; long t = idx / Wo;
    int ho = t % Ho;   t /= Ho;
    int co = t % Co;   int b = (int)(t / Co);

    float acc = bias[co];
    for (int ci = 0; ci < Ci; ++ci) {
        const float* ip = in + ((long)(b * Ci + ci) * Hi) * Wi;
        const float* wp = w + (long)(co * Ci + ci) * 9;
        #pragma unroll
        for (int kh = 0; kh < 3; ++kh) {
            int hi = ho * sh + kh - 1;
            if (hi < 0 || hi >= Hi) continue;
            #pragma unroll
            for (int kw = 0; kw < 3; ++kw) {
                int wi = wo * sw + kw - 1;
                if (wi < 0 || wi >= Wi) continue;
                acc += ip[(long)hi * Wi + wi] * wp[kh * 3 + kw];
            }
        }
    }
    acc = (acc - mu[co]) * (g[co] * rsqrtf(var[co] + BN_EPS)) + be[co];
    if (relu) acc = fmaxf(acc, 0.f);
    out[idx] = acc;
}

// ---------------------------------------------------------------------------
// x1 [B,32,24,24] -> ht[b,h,c*24+w], hf[b,w,c*24+h]   (both [B,24,768])
// ---------------------------------------------------------------------------
__global__ void nodebuild_kernel(const float* __restrict__ x1,
                                 float* __restrict__ ht, float* __restrict__ hf)
{
    long idx = (long)blockIdx.x * blockDim.x + threadIdx.x;
    long total = (long)BATCH * 32 * 24 * 24;
    if (idx >= total) return;
    int ww = idx % 24; long t = idx / 24;
    int hh = t % 24;   t /= 24;
    int c  = t % 32;   int b = (int)(t / 32);
    float v = x1[idx];
    ht[((long)b * 24 + hh) * 768 + c * 24 + ww] = v;
    hf[((long)b * 24 + ww) * 768 + c * 24 + hh] = v;
}

// ---------------------------------------------------------------------------
// Per-sample cosine top-2 graph -> normalized adjacency An = D^-1/2 (A+I) D^-1/2
// ---------------------------------------------------------------------------
__global__ void graph_kernel(const float* __restrict__ Hn, float* __restrict__ An)
{
    __shared__ float nrm[24];
    __shared__ float sim[576];
    __shared__ int   i1[24], i2[24];
    __shared__ float dinv[24];
    const int b = blockIdx.x;
    const float* h = Hn + (long)b * 24 * 768;
    const int tid = threadIdx.x;

    if (tid < 24) {
        float s = 0.f;
        for (int d = 0; d < 768; ++d) { float v = h[tid * 768 + d]; s += v * v; }
        nrm[tid] = fmaxf(sqrtf(s), 1e-12f);
    }
    __syncthreads();
    for (int p = tid; p < 576; p += blockDim.x) {
        int i = p / 24, j = p % 24;
        float s = 0.f;
        for (int d = 0; d < 768; ++d) s += h[i * 768 + d] * h[j * 768 + d];
        sim[p] = (i == j) ? -3.0e38f : s / (nrm[i] * nrm[j]);
    }
    __syncthreads();
    if (tid < 24) {
        int b1 = 0, b2 = 0; float v1 = -3.9e38f, v2 = -3.9e38f;
        for (int j = 0; j < 24; ++j) {
            float v = sim[tid * 24 + j];
            if (v > v1) { v1 = v; b1 = j; }
        }
        for (int j = 0; j < 24; ++j) {
            if (j == b1) continue;
            float v = sim[tid * 24 + j];
            if (v > v2) { v2 = v; b2 = j; }
        }
        i1[tid] = b1; i2[tid] = b2;
        float deg = 1.f;
        if (b1 != tid) deg += 1.f;
        if (b2 != tid && b2 != b1) deg += 1.f;
        dinv[tid] = rsqrtf(deg);
    }
    __syncthreads();
    for (int p = tid; p < 576; p += blockDim.x) {
        int i = p / 24, j = p % 24;
        float a = (i == j || j == i1[i] || j == i2[i]) ? 1.f : 0.f;
        An[(long)b * 576 + p] = a * dinv[i] * dinv[j];
    }
}

// ---------------------------------------------------------------------------
// GCN aggregation (z = An @ XW + gb, relu) + grouped 1x1 conv (+c1b) + BN1d
// ---------------------------------------------------------------------------
__global__ void gcn_post_kernel(const float* __restrict__ XW, const float* __restrict__ An,
                                const float* __restrict__ gb, const float* __restrict__ c1w,
                                const float* __restrict__ c1b,
                                const float* __restrict__ g, const float* __restrict__ be,
                                const float* __restrict__ mu, const float* __restrict__ var,
                                float* __restrict__ out, int fout)
{
    const int G = fout >> 2;
    long idx = (long)blockIdx.x * blockDim.x + threadIdx.x;
    long total = (long)BATCH * 24 * G;
    if (idx >= total) return;
    int gi = idx % G; long t = idx / G;
    int n = t % 24; int b = (int)(t / 24);
    const float* an  = An + (long)b * 576 + n * 24;
    const float* xwb = XW + (long)b * 24 * fout;

    float z[4];
    #pragma unroll
    for (int j = 0; j < 4; ++j) {
        int d = gi * 4 + j;
        float s = gb[d];
        for (int ss = 0; ss < 24; ++ss) s += an[ss] * xwb[(long)ss * fout + d];
        z[j] = fmaxf(s, 0.f);
    }
    #pragma unroll
    for (int o = 0; o < 4; ++o) {
        int d = gi * 4 + o;
        float s = c1b[d];
        #pragma unroll
        for (int j = 0; j < 4; ++j) s += z[j] * c1w[d * 4 + j];
        s = (s - mu[d]) * (g[d] * rsqrtf(var[d] + BN_EPS)) + be[d];
        out[((long)b * 24 + n) * fout + d] = s;
    }
}

// ---------------------------------------------------------------------------
// concat:  y[b, n*24+d] = Ht[(b,n),d] ;  y[b, 576 + n*24+d] = Hf[(b,n),d]
// ---------------------------------------------------------------------------
__global__ void concat_kernel(const float* __restrict__ Ht, const float* __restrict__ Hf,
                              float* __restrict__ y)
{
    long idx = (long)blockIdx.x * blockDim.x + threadIdx.x;
    long total = (long)BATCH * 1152;
    if (idx >= total) return;
    int r = idx % 1152; int b = (int)(idx / 1152);
    const float* src = (r < 576) ? Ht : Hf;
    int rr = (r < 576) ? r : r - 576;
    int n = rr / 24, d = rr % 24;
    y[idx] = src[((long)b * 24 + n) * 24 + d];
}

// ===========================================================================
// Host side
// ===========================================================================
static inline void gemm(const float* A, const float* W, float* C, int M, int N, int K,
                        const float* bias, const float* bng, const float* bnb,
                        const float* bnm, const float* bnv,
                        const float* resid, int relu, hipStream_t s)
{
    dim3 grid((N + 15) / 16, M / 64);   // M is always a multiple of 64 here
    wmma_gemm_kernel<<<grid, 32, 0, s>>>(A, W, C, M, N, K,
                                         bias, bng, bnb, bnm, bnv, resid, relu);
}

static void run_encoder(void* const* d_in, int base, float* P0 /*3072x768, holds h*/,
                        float* P1 /*3072x384*/, const float* An,
                        float* XW, float* Zb, float* Fb, hipStream_t s)
{
    auto in = [&](int i) { return (const float*)d_in[i]; };
    const int M = BATCH * 24;
    float* cur = P0;
    for (int l = 0; l < 5; ++l) {
        int fin = 768 >> l, fout = fin >> 1;
        int o = base + 22 * l;
        const float *gw = in(o + 0), *gb = in(o + 1), *c1w = in(o + 2), *c1b = in(o + 3);
        const float *b1g = in(o + 4), *b1b = in(o + 5), *b1m = in(o + 6), *b1v = in(o + 7);
        const float *rw = in(o + 8), *rb = in(o + 9);
        const float *f1w = in(o + 10), *f1b = in(o + 11);
        const float *fb1g = in(o + 12), *fb1b = in(o + 13), *fb1m = in(o + 14), *fb1v = in(o + 15);
        const float *f2w = in(o + 16), *f2b = in(o + 17);
        const float *fb2g = in(o + 18), *fb2b = in(o + 19), *fb2m = in(o + 20), *fb2v = in(o + 21);
        float* nxt = (l & 1) ? P0 : P1;

        // XW = H @ gw (no epilogue)
        gemm(cur, gw, XW, M, fout, fin, nullptr, nullptr, nullptr, nullptr, nullptr, nullptr, 0, s);
        // Zb = BN1( groupedconv( relu(An@XW + gb) ) + c1b )
        {
            long tot = (long)M * (fout >> 2);
            gcn_post_kernel<<<(tot + 255) / 256, 256, 0, s>>>(XW, An, gb, c1w, c1b,
                                                              b1g, b1b, b1m, b1v, Zb, fout);
        }
        // Zb = Zb + (H @ rw + rb)   (alias safe: each thread reads its own element first)
        gemm(cur, rw, Zb, M, fout, fin, rb, nullptr, nullptr, nullptr, nullptr, Zb, 0, s);
        // Fb = relu(BN(Zb @ f1w + f1b))
        gemm(Zb, f1w, Fb, M, fout, fout, f1b, fb1g, fb1b, fb1m, fb1v, nullptr, 1, s);
        // Hnext = BN(Fb @ f2w + f2b) + Zb
        gemm(Fb, f2w, nxt, M, fout, fout, f2b, fb2g, fb2b, fb2m, fb2v, Zb, 0, s);
        cur = nxt;
    }
    // after 5 layers the final h (3072 x 24) lives in P1
}

extern "C" void kernel_launch(void* const* d_in, const int* in_sizes, int n_in,
                              void* d_out, int out_size, void* d_ws, size_t ws_size,
                              hipStream_t stream)
{
    (void)in_sizes; (void)n_in; (void)out_size; (void)ws_size;
    auto in = [&](int i) { return (const float*)d_in[i]; };

    // ---- workspace layout (floats) ----
    float* ws = (float*)d_ws;
    long off = 0;
    float* buf0 = ws + off; off += (long)BATCH * 8  * 192 * 216;  // slice in, c2/c4 out
    float* buf1 = ws + off; off += (long)BATCH * 16 * 192 * 216;  // c1/c3 out
    float* x1   = ws + off; off += (long)BATCH * 32 * 24 * 24;
    float* h_t  = ws + off; off += (long)BATCH * 24 * 768;        // P0 for enc_t
    float* h_f  = ws + off; off += (long)BATCH * 24 * 768;        // P0 for enc_f
    float* An_t = ws + off; off += (long)BATCH * 576;
    float* An_f = ws + off; off += (long)BATCH * 576;
    float* eh_t = ws + off; off += (long)BATCH * 24 * 384;        // P1 for enc_t
    float* eh_f = ws + off; off += (long)BATCH * 24 * 384;        // P1 for enc_f
    float* XW   = ws + off; off += (long)BATCH * 24 * 384;
    float* Zb   = ws + off; off += (long)BATCH * 24 * 384;
    float* Fb   = ws + off; off += (long)BATCH * 24 * 384;
    float* ycat = ws + off; off += (long)BATCH * 1152;
    float* y1   = ws + off; off += (long)BATCH * 256;
    float* y2   = ws + off; off += (long)BATCH * 128;

    // ---- stem ----
    {
        long tot = (long)BATCH * 384 * 216;
        slice_kernel<<<(tot + 255) / 256, 256, 0, stream>>>(in(0), buf0);
    }
    struct CL { int Ci, Co, Hi, Wi, Ho, Wo, sh, sw, relu; const float *src; float *dst; };
    const CL convs[5] = {
        { 1,  4, 384, 216, 384, 216, 1, 1, 1, buf0, buf1 },
        { 4,  8, 384, 216, 192, 216, 2, 1, 1, buf1, buf0 },
        { 8, 16, 192, 216, 192, 216, 1, 1, 1, buf0, buf1 },
        {16, 32, 192, 216,  96,  72, 2, 3, 1, buf1, buf0 },
        {32, 32,  96,  72,  24,  24, 4, 3, 0, buf0, x1   },
    };
    for (int i = 0; i < 5; ++i) {
        const CL& c = convs[i];
        long tot = (long)BATCH * c.Co * c.Ho * c.Wo;
        conv_bn_kernel<<<(tot + 255) / 256, 256, 0, stream>>>(
            c.src, c.dst, in(1 + 2 * i), in(2 + 2 * i),
            in(11 + 4 * i), in(12 + 4 * i), in(13 + 4 * i), in(14 + 4 * i),
            c.Ci, c.Co, c.Hi, c.Wi, c.Ho, c.Wo, c.sh, c.sw, c.relu);
    }

    // ---- nodes + graphs ----
    {
        long tot = (long)BATCH * 32 * 576;
        nodebuild_kernel<<<(tot + 255) / 256, 256, 0, stream>>>(x1, h_t, h_f);
    }
    graph_kernel<<<BATCH, 192, 0, stream>>>(h_t, An_t);
    graph_kernel<<<BATCH, 192, 0, stream>>>(h_f, An_f);

    // ---- encoders (graphs built before encoders overwrite h_t/h_f) ----
    run_encoder(d_in,  31, h_t, eh_t, An_t, XW, Zb, Fb, stream);
    run_encoder(d_in, 141, h_f, eh_f, An_f, XW, Zb, Fb, stream);

    // ---- head ----
    {
        long tot = (long)BATCH * 1152;
        concat_kernel<<<(tot + 255) / 256, 256, 0, stream>>>(eh_t, eh_f, ycat);
    }
    const int F = 251;
    gemm(ycat, in(F + 0), y1, BATCH, 256, 1152, in(F + 1),
         nullptr, nullptr, nullptr, nullptr, nullptr, 1, stream);
    gemm(y1,   in(F + 2), y2, BATCH, 128, 256,  in(F + 3),
         nullptr, nullptr, nullptr, nullptr, nullptr, 1, stream);
    gemm(y2,   in(F + 4), (float*)d_out, BATCH, 80, 128, in(F + 5),
         nullptr, nullptr, nullptr, nullptr, nullptr, 0, stream);
}
